// SelfAttLayer_34926674051619
// MI455X (gfx1250) — matrix operand order repaired and checked by reading, not compiled, not run on hardware
//
#include <hip/hip_runtime.h>
#include <hip/hip_bf16.h>

// ---------------------------------------------------------------------------
// GraphTransformer attention layer for MI455X (gfx1250, wave32, WMMA).
// Edge MLPs dominate (~472 GFLOP) -> bf16 v_wmma_f32_16x16x32_bf16 GEMMs.
// M-tiled 4x (64 edges/block) so each pre-swizzled B-fragment load feeds 4
// WMMAs -> ~4x less L2 weight traffic. Activations overlay the input tile in
// LDS; logits come straight out of C-fragments via 16-lane shfl reductions.
// ---------------------------------------------------------------------------

typedef __bf16 v16bf __attribute__((ext_vector_type(16)));
typedef __bf16 v8bf  __attribute__((ext_vector_type(8)));
typedef float  v8f   __attribute__((ext_vector_type(8)));
typedef float  v4f   __attribute__((ext_vector_type(4)));

#define NNODES 50000
#define NEDGES 800000
#define IN_DIM 128
#define HID    256
#define OUTD   256
#define NHEAD  8
#define KVIN   320   // e(64) | h[dst](128) | h[src](128)
#define MT     4     // M-tiles (of 16 rows) per block in edge kernels

__device__ __forceinline__ __bf16 f2bf(float f) {
  union { float f; unsigned u; } a; a.f = f;
  unsigned u = a.u;
  u += 0x7FFFu + ((u >> 16) & 1u);          // round-to-nearest-even
  union { unsigned short s; __bf16 b; } r;
  r.s = (unsigned short)(u >> 16);
  return r.b;
}

// WMMA A-fragment (16x32 bf16) from an LDS tile, per CDNA5 ISA 7.12.2:
// lanes 0-15 : M=lane,    elems 0-7 -> K=kc*32+0..7,  elems 8-15 -> K=kc*32+16..23
// lanes 16-31: M=lane-16, elems 0-7 -> K=kc*32+8..15, elems 8-15 -> K=kc*32+24..31
__device__ __forceinline__ v16bf a_frag(const __bf16* lds, int stride, int kc, int lane) {
  const int m  = lane & 15;
  const int kb = kc * 32 + ((lane & 16) ? 8 : 0);
  const __bf16* p = lds + m * stride + kb;
  v8bf lo = *(const v8bf*)(p);
  v8bf hi = *(const v8bf*)(p + 16);
  return __builtin_shufflevector(lo, hi, 0,1,2,3,4,5,6,7,8,9,10,11,12,13,14,15);
}

// B-fragments are pre-swizzled by prep_weights -> single contiguous 32B load.
__device__ __forceinline__ v16bf b_frag(const __bf16* w, int kc, int nt, int NT, int lane) {
  return *(const v16bf*)(w + (size_t)(((kc * NT) + nt) * 32 + lane) * 16);
}

__device__ __forceinline__ v8f wmma_bf16(v16bf a, v16bf b, v8f c) {
  return __builtin_amdgcn_wmma_f32_16x16x32_bf16(false, a, false, b, (short)0, c, false, false);
}

__device__ __forceinline__ void atomicMaxFloat(float* addr, float val) {
  // ordered-int trick (monotonic for IEEE754): works for mixed signs
  if (val >= 0.f) atomicMax((int*)addr, __float_as_int(val));
  else            atomicMin((unsigned int*)addr, __float_as_uint(val));
}

// --------------------------- weight pre-swizzle ----------------------------
// Pack W[K][N] (row-major f32) into WMMA B-fragment order (bf16):
// frag[kc][nt][lane][e] = W[kc*32 + ((lane&16)?8:0) + (e&7) + ((e&8)?16:0)][nt*16 + (lane&15)]
__global__ void prep_weights(const float* __restrict__ W, __bf16* __restrict__ out,
                             int K, int N) {
  const int NT = N >> 4;
  const int total = K * N;
  int idx = blockIdx.x * blockDim.x + threadIdx.x;
  if (idx >= total) return;
  int e    = idx & 15;
  int lane = (idx >> 4) & 31;
  int nt   = (idx >> 9) % NT;
  int kc   = idx / (512 * NT);
  int n = nt * 16 + (lane & 15);
  int k = kc * 32 + ((lane & 16) ? 8 : 0) + (e & 7) + ((e & 8) ? 16 : 0);
  out[idx] = f2bf(W[(size_t)k * N + n]);
}

__global__ void fill_f32(float* __restrict__ p, float v, int n) {
  int i = blockIdx.x * blockDim.x + threadIdx.x;
  if (i < n) p[i] = v;
}

// ------------------------------- Q = MLP(h) --------------------------------
__global__ __launch_bounds__(128) void q_kernel(
    const float* __restrict__ h,
    const __bf16* __restrict__ w1, const float* __restrict__ b1,
    const __bf16* __restrict__ w2, const float* __restrict__ b2,
    float* __restrict__ qout) {
  __shared__ __attribute__((aligned(16))) __bf16 xin[16 * IN_DIM];
  __shared__ __attribute__((aligned(16))) __bf16 act[16 * HID];
  const int tid  = threadIdx.x;
  const int lane = tid & 31;
  const int wave = tid >> 5;
  const int node0 = blockIdx.x * 16;

  // gather 16x128 f32 -> bf16 LDS (512 float4, 4 per thread)
  #pragma unroll
  for (int i = 0; i < 4; ++i) {
    int idx = tid + 128 * i;
    int row = idx >> 5;
    int c4  = (idx & 31) * 4;
    v4f v = *(const v4f*)(h + (size_t)(node0 + row) * IN_DIM + c4);
    __bf16* d = xin + row * IN_DIM + c4;
    d[0] = f2bf(v.x); d[1] = f2bf(v.y); d[2] = f2bf(v.z); d[3] = f2bf(v.w);
  }
  __syncthreads();

  const int mbase = (lane & 16) ? 8 : 0;
  const int nlo   = lane & 15;

  v8f acc[4] = {};
  #pragma unroll
  for (int kc = 0; kc < IN_DIM / 32; ++kc) {
    v16bf a = a_frag(xin, IN_DIM, kc, lane);
    #pragma unroll
    for (int j = 0; j < 4; ++j)
      acc[j] = wmma_bf16(a, b_frag(w1, kc, wave * 4 + j, HID / 16, lane), acc[j]);
  }
  #pragma unroll
  for (int j = 0; j < 4; ++j) {
    int col = (wave * 4 + j) * 16 + nlo;
    float bb = b1[col];
    #pragma unroll
    for (int i = 0; i < 8; ++i) {
      float x = acc[j][i] + bb;
      act[(mbase + i) * HID + col] = f2bf(x > 0.f ? x : 0.f);
    }
  }
  __syncthreads();

  v8f acc2[4] = {};
  #pragma unroll
  for (int kc = 0; kc < HID / 32; ++kc) {
    v16bf a = a_frag(act, HID, kc, lane);
    #pragma unroll
    for (int j = 0; j < 4; ++j)
      acc2[j] = wmma_bf16(a, b_frag(w2, kc, wave * 4 + j, OUTD / 16, lane), acc2[j]);
  }
  #pragma unroll
  for (int j = 0; j < 4; ++j) {
    int col = (wave * 4 + j) * 16 + nlo;
    float bb = b2[col];
    #pragma unroll
    for (int i = 0; i < 8; ++i)
      qout[(size_t)(node0 + mbase + i) * OUTD + col] = acc2[j][i] + bb;
  }
}

// ------------- shared edge-MLP body: kv gather + 2 WMMA layers -------------
// smem overlays: [64 x 320 bf16] kv input, then [64 x 256 bf16] activations.
__device__ __forceinline__ void edge_mlp(
    __bf16* smem, const int* esrc, const int* edst,
    const float* __restrict__ eattr, const float* __restrict__ h,
    const __bf16* __restrict__ w1, const float* __restrict__ b1,
    const __bf16* __restrict__ w2,
    int e0, int tid, int lane, int wave, int mbase, int nlo,
    v8f (&acc2)[MT][4]) {
  // gather [e | h[dst] | h[src]] -> 64x320 bf16 (5120 float4, 40/thread)
  #pragma unroll 4
  for (int i = 0; i < 40; ++i) {
    int idx = tid + 128 * i;
    int row = idx / 80;
    int g   = idx - row * 80;
    v4f v; int col;
    if (g < 16)      { v = *(const v4f*)(eattr + (size_t)(e0 + row) * 64 + g * 4);          col = g * 4; }
    else if (g < 48) { v = *(const v4f*)(h + (size_t)edst[row] * IN_DIM + (g - 16) * 4);    col = 64 + (g - 16) * 4; }
    else             { v = *(const v4f*)(h + (size_t)esrc[row] * IN_DIM + (g - 48) * 4);    col = 192 + (g - 48) * 4; }
    __bf16* d = smem + row * KVIN + col;
    d[0] = f2bf(v.x); d[1] = f2bf(v.y); d[2] = f2bf(v.z); d[3] = f2bf(v.w);
  }
  __syncthreads();

  // layer 1: [64x320] @ [320x256], B-frag reused across 4 M-tiles
  v8f acc[MT][4] = {};
  #pragma unroll
  for (int kc = 0; kc < KVIN / 32; ++kc) {
    v16bf a[MT];
    #pragma unroll
    for (int mt = 0; mt < MT; ++mt)
      a[mt] = a_frag(smem + mt * 16 * KVIN, KVIN, kc, lane);
    #pragma unroll
    for (int j = 0; j < 4; ++j) {
      v16bf b = b_frag(w1, kc, wave * 4 + j, HID / 16, lane);
      #pragma unroll
      for (int mt = 0; mt < MT; ++mt)
        acc[mt][j] = wmma_bf16(a[mt], b, acc[mt][j]);
    }
  }
  __syncthreads();            // kv reads done -> safe to overlay activations

  #pragma unroll
  for (int mt = 0; mt < MT; ++mt)
    #pragma unroll
    for (int j = 0; j < 4; ++j) {
      int col = (wave * 4 + j) * 16 + nlo;
      float bb = b1[col];
      #pragma unroll
      for (int i = 0; i < 8; ++i) {
        float x = acc[mt][j][i] + bb;
        smem[(mt * 16 + mbase + i) * HID + col] = f2bf(x > 0.f ? x : 0.f);
      }
    }
  __syncthreads();

  // layer 2: [64x256] @ [256x256]
  #pragma unroll
  for (int kc = 0; kc < HID / 32; ++kc) {
    v16bf a[MT];
    #pragma unroll
    for (int mt = 0; mt < MT; ++mt)
      a[mt] = a_frag(smem + mt * 16 * HID, HID, kc, lane);
    #pragma unroll
    for (int j = 0; j < 4; ++j) {
      v16bf b = b_frag(w2, kc, wave * 4 + j, OUTD / 16, lane);
      #pragma unroll
      for (int mt = 0; mt < MT; ++mt)
        acc2[mt][j] = wmma_bf16(a[mt], b, acc2[mt][j]);
    }
  }
}

// --------------------- Edge pass 1: K MLP + logits + max --------------------
__global__ __launch_bounds__(128) void edge_k_kernel(
    const float* __restrict__ eattr, const float* __restrict__ h,
    const long long* __restrict__ eidx,
    const __bf16* __restrict__ w1, const float* __restrict__ b1,
    const __bf16* __restrict__ w2, const float* __restrict__ b2,
    const float* __restrict__ q, float* __restrict__ logits,
    float* __restrict__ mx) {
  __shared__ __attribute__((aligned(16))) __bf16 smem[64 * KVIN];
  __shared__ int esrc[64], edst[64];
  const int tid  = threadIdx.x;
  const int lane = tid & 31;
  const int wave = tid >> 5;
  const int e0 = blockIdx.x * (16 * MT);
  const int mbase = (lane & 16) ? 8 : 0;
  const int nlo   = lane & 15;

  if (tid < 64) {
    esrc[tid] = (int)eidx[e0 + tid];
    edst[tid] = (int)eidx[NEDGES + e0 + tid];
  }
  __syncthreads();

  v8f acc2[MT][4] = {};
  edge_mlp(smem, esrc, edst, eattr, h, w1, b1, w2,
           e0, tid, lane, wave, mbase, nlo, acc2);

  // ---- logits straight from C-fragments ----
  // wave owns cols [wave*64, wave*64+63] = heads 2*wave, 2*wave+1.
  // lane holds col = wave*64 + j*16 + nlo; reduce over the 16-lane half.
  float bb[4];
  #pragma unroll
  for (int j = 0; j < 4; ++j) bb[j] = b2[wave * 64 + j * 16 + nlo];
  const int h0 = 2 * wave, h1 = 2 * wave + 1;

  #pragma unroll
  for (int mt = 0; mt < MT; ++mt) {
    #pragma unroll
    for (int i = 0; i < 8; ++i) {
      int r = mt * 16 + mbase + i;             // same for the whole 16-lane half
      int d = edst[r];
      const float* qrow = q + (size_t)d * OUTD + wave * 64 + nlo;
      float p0 = (acc2[mt][0][i] + bb[0]) * qrow[0]  + (acc2[mt][1][i] + bb[1]) * qrow[16];
      float p1 = (acc2[mt][2][i] + bb[2]) * qrow[32] + (acc2[mt][3][i] + bb[3]) * qrow[48];
      #pragma unroll
      for (int off = 8; off >= 1; off >>= 1) {  // stays within the 16-lane half
        p0 += __shfl_xor(p0, off, 32);
        p1 += __shfl_xor(p1, off, 32);
      }
      if (nlo == 0) {
        float l0 = p0 * 0.17677669529663687f;  // 1/sqrt(32)
        float l1 = p1 * 0.17677669529663687f;
        logits[(size_t)(e0 + r) * NHEAD + h0] = l0;
        logits[(size_t)(e0 + r) * NHEAD + h1] = l1;
        atomicMaxFloat(mx + (size_t)d * NHEAD + h0, l0);
        atomicMaxFloat(mx + (size_t)d * NHEAD + h1, l1);
      }
    }
  }
}

// --------------------- exp + denominator scatter-sum -----------------------
__global__ void ex_kernel(float* __restrict__ lg, const float* __restrict__ mx,
                          float* __restrict__ den, const long long* __restrict__ eidx) {
  int i = blockIdx.x * blockDim.x + threadIdx.x;
  if (i >= NEDGES * NHEAD) return;
  int e = i >> 3, hd = i & 7;
  long long d = eidx[NEDGES + e];
  float ex = __expf(lg[i] - mx[d * NHEAD + hd]);
  lg[i] = ex;
  atomicAdd(den + d * NHEAD + hd, ex);
}

// --------------- Edge pass 2: V MLP + alpha-weighted scatter ---------------
__global__ __launch_bounds__(128) void edge_v_kernel(
    const float* __restrict__ eattr, const float* __restrict__ h,
    const long long* __restrict__ eidx,
    const __bf16* __restrict__ w1, const float* __restrict__ b1,
    const __bf16* __restrict__ w2, const float* __restrict__ b2,
    const float* __restrict__ ex, const float* __restrict__ den,
    float* __restrict__ out) {
  __shared__ __attribute__((aligned(16))) __bf16 smem[64 * KVIN];
  __shared__ float al[64 * NHEAD];
  __shared__ int esrc[64], edst[64];
  const int tid  = threadIdx.x;
  const int lane = tid & 31;
  const int wave = tid >> 5;
  const int e0 = blockIdx.x * (16 * MT);
  const int mbase = (lane & 16) ? 8 : 0;
  const int nlo   = lane & 15;

  if (tid < 64) {
    esrc[tid] = (int)eidx[e0 + tid];
    edst[tid] = (int)eidx[NEDGES + e0 + tid];
  }
  __syncthreads();

  // alpha = ex / den[dst]  (64x8 = 512 values, 4 per thread)
  #pragma unroll
  for (int i = 0; i < 4; ++i) {
    int idx = tid + 128 * i;
    int r = idx >> 3, hd = idx & 7;
    al[idx] = ex[(size_t)(e0 + r) * NHEAD + hd] /
              den[(size_t)edst[r] * NHEAD + hd];
  }

  v8f acc2[MT][4] = {};
  edge_mlp(smem, esrc, edst, eattr, h, w1, b1, w2,
           e0, tid, lane, wave, mbase, nlo, acc2);

  // weighted scatter: out[dst] += alpha * v
  #pragma unroll
  for (int mt = 0; mt < MT; ++mt) {
    #pragma unroll
    for (int j = 0; j < 4; ++j) {
      int col = wave * 64 + j * 16 + nlo;
      float bb = b2[col];
      int hd = col >> 5;
      #pragma unroll
      for (int i = 0; i < 8; ++i) {
        int r = mt * 16 + mbase + i;
        float v = (acc2[mt][j][i] + bb) * al[r * NHEAD + hd];
        atomicAdd(out + (size_t)edst[r] * OUTD + col, v);
      }
    }
  }
}

// ---------------------------------------------------------------------------
extern "C" void kernel_launch(void* const* d_in, const int* in_sizes, int n_in,
                              void* d_out, int out_size, void* d_ws, size_t ws_size,
                              hipStream_t stream) {
  (void)in_sizes; (void)n_in; (void)out_size; (void)ws_size;
  const float*     h   = (const float*)d_in[0];
  const float*     e   = (const float*)d_in[1];
  const long long* ei  = (const long long*)d_in[2];
  const float* Wk1 = (const float*)d_in[3];  const float* bk1 = (const float*)d_in[4];
  const float* Wk2 = (const float*)d_in[5];  const float* bk2 = (const float*)d_in[6];
  const float* Wv1 = (const float*)d_in[7];  const float* bv1 = (const float*)d_in[8];
  const float* Wv2 = (const float*)d_in[9];  const float* bv2 = (const float*)d_in[10];
  const float* Wq1 = (const float*)d_in[11]; const float* bq1 = (const float*)d_in[12];
  const float* Wq2 = (const float*)d_in[13]; const float* bq2 = (const float*)d_in[14];

  char* ws = (char*)d_ws;
  __bf16* wk1 = (__bf16*)(ws + 0);          // 320x256 bf16 = 163840 B
  __bf16* wk2 = (__bf16*)(ws + 163840);     // 256x256 bf16 = 131072 B
  __bf16* wv1 = (__bf16*)(ws + 294912);     // 163840 B
  __bf16* wv2 = (__bf16*)(ws + 458752);     // 131072 B
  __bf16* wq1 = (__bf16*)(ws + 589824);     // 128x256 bf16 = 65536 B
  __bf16* wq2 = (__bf16*)(ws + 655360);     // 131072 B -> end 786432
  float* qbuf = (float*)(ws + 786432);                          // N*256 f32 = 51.2 MB
  float* lg   = (float*)(ws + 786432 + 51200000);               // E*8  f32 = 25.6 MB
  float* mx   = (float*)(ws + 786432 + 51200000 + 25600000);    // N*8  f32 = 1.6 MB
  float* den  = (float*)(ws + 786432 + 51200000 + 25600000 + 1600000);
  float* out  = (float*)d_out;

  // 1) weights -> bf16 B-fragment order
  prep_weights<<<(320 * 256 + 255) / 256, 256, 0, stream>>>(Wk1, wk1, KVIN, HID);
  prep_weights<<<(256 * 256 + 255) / 256, 256, 0, stream>>>(Wk2, wk2, HID, OUTD);
  prep_weights<<<(320 * 256 + 255) / 256, 256, 0, stream>>>(Wv1, wv1, KVIN, HID);
  prep_weights<<<(256 * 256 + 255) / 256, 256, 0, stream>>>(Wv2, wv2, HID, OUTD);
  prep_weights<<<(128 * 256 + 255) / 256, 256, 0, stream>>>(Wq1, wq1, IN_DIM, HID);
  prep_weights<<<(256 * 256 + 255) / 256, 256, 0, stream>>>(Wq2, wq2, HID, OUTD);

  // 2) init accumulators
  fill_f32<<<(NNODES * NHEAD + 255) / 256, 256, 0, stream>>>(mx, -3.0e38f, NNODES * NHEAD);
  fill_f32<<<(NNODES * NHEAD + 255) / 256, 256, 0, stream>>>(den, 0.f, NNODES * NHEAD);
  fill_f32<<<(NNODES * OUTD + 255) / 256, 256, 0, stream>>>(out, 0.f, NNODES * OUTD);

  // 3) Q = MLP(h)
  q_kernel<<<NNODES / 16, 128, 0, stream>>>(h, wq1, bq1, wq2, bq2, qbuf);

  // 4) K MLP + logits + segment max
  edge_k_kernel<<<NEDGES / (16 * MT), 128, 0, stream>>>(e, h, ei, wk1, bk1, wk2, bk2,
                                                        qbuf, lg, mx);
  // 5) exp + denominator
  ex_kernel<<<(NEDGES * NHEAD + 255) / 256, 256, 0, stream>>>(lg, mx, den, ei);

  // 6) V MLP + alpha-weighted scatter-sum
  edge_v_kernel<<<NEDGES / (16 * MT), 128, 0, stream>>>(e, h, ei, wv1, bv1, wv2, bv2,
                                                        lg, den, out);
}
